// RRWPLinearEdgeMaskedEncoder_70849780514913
// MI455X (gfx1250) — matrix operand order repaired and checked by reading, not compiled, not run on hardware
//
#include <hip/hip_runtime.h>
#include <stdint.h>

#define N_NODES 40000
#define OUT_DIM 128
#define EMB_DIM 32

#define EMIT_TPB 256
#define WPT 8
#define GEMM_TPB 256

typedef __attribute__((ext_vector_type(16))) _Float16 v16h;
typedef __attribute__((ext_vector_type(8)))  _Float16 v8h;
typedef __attribute__((ext_vector_type(8)))  float    v8f;

static __device__ __forceinline__ int bsearch_key(const unsigned* __restrict__ a,
                                                  int n, unsigned key) {
  int lo = 0, hi = n;
  while (lo < hi) {
    int mid = (lo + hi) >> 1;
    if (a[mid] < key) lo = mid + 1; else hi = mid;
  }
  return (lo < n && a[lo] == key) ? lo : -1;
}

// ---------------- zero kernels ----------------
__global__ void k_zero_u4(uint4* p, long long n4) {
  long long i = (long long)blockIdx.x * blockDim.x + threadIdx.x;
  if (i < n4) p[i] = make_uint4(0u, 0u, 0u, 0u);
}

__global__ void k_zero_f32(float* p, long long n) {
  long long i = (long long)blockIdx.x * blockDim.x + threadIdx.x;
  if (i < n) p[i] = 0.0f;
}

// ---------------- bitmap set ----------------
// mask keys = all edge keys (self-loop edges harmlessly duplicate loop keys)
//           + all N self-loop keys
__global__ void k_set_bits(const int* __restrict__ e0, const int* __restrict__ e1,
                           unsigned* __restrict__ bitmap, int E, int N) {
  int i = blockIdx.x * blockDim.x + threadIdx.x;
  int M = E + N;
  if (i >= M) return;
  unsigned key;
  if (i < E) key = (unsigned)e0[i] * (unsigned)N + (unsigned)e1[i];
  else { unsigned r = (unsigned)(i - E); key = r * (unsigned)N + r; }
  atomicOr(&bitmap[key >> 5], 1u << (key & 31u));
}

// ---------------- per-block popcount ----------------
__global__ void k_blockpop(const unsigned* __restrict__ bitmap,
                           unsigned* __restrict__ blockSums, int NW) {
  __shared__ unsigned s[EMIT_TPB];
  int tid = threadIdx.x;
  int base = blockIdx.x * (EMIT_TPB * WPT) + tid * WPT;
  unsigned c = 0;
#pragma unroll
  for (int j = 0; j < WPT; ++j) {
    int w = base + j;
    if (w < NW) c += __popc(bitmap[w]);
  }
  s[tid] = c;
  __syncthreads();
  for (int off = EMIT_TPB / 2; off > 0; off >>= 1) {
    if (tid < off) s[tid] += s[tid + off];
    __syncthreads();
  }
  if (tid == 0) blockSums[blockIdx.x] = s[0];
}

// ---------------- single-workgroup exclusive scan (in place) ----------------
__global__ void k_scan_single(unsigned* a, int n) {
  __shared__ unsigned s[1024];
  int tid = threadIdx.x;
  unsigned carry = 0;
  for (int base = 0; base < n; base += 1024) {
    int i = base + tid;
    unsigned v = (i < n) ? a[i] : 0u;
    s[tid] = v;
    __syncthreads();
    for (int off = 1; off < 1024; off <<= 1) {
      unsigned t = (tid >= off) ? s[tid - off] : 0u;
      __syncthreads();
      s[tid] += t;
      __syncthreads();
    }
    if (i < n) a[i] = s[tid] - v + carry;  // exclusive
    unsigned total = s[1023];
    __syncthreads();
    carry += total;
  }
}

// ---------------- emit sorted unique keys + out_idx ----------------
__global__ void k_emit(const unsigned* __restrict__ bitmap,
                       const unsigned* __restrict__ blockOffs,
                       unsigned* __restrict__ mkeys,
                       float* __restrict__ out_idx, int U, int NW, int N) {
  __shared__ unsigned s[EMIT_TPB];
  int tid = threadIdx.x;
  int base = blockIdx.x * (EMIT_TPB * WPT) + tid * WPT;
  unsigned w[WPT];
  unsigned c = 0;
#pragma unroll
  for (int j = 0; j < WPT; ++j) {
    int wi = base + j;
    w[j] = (wi < NW) ? bitmap[wi] : 0u;
    c += __popc(w[j]);
  }
  s[tid] = c;
  __syncthreads();
  for (int off = 1; off < EMIT_TPB; off <<= 1) {
    unsigned t = (tid >= off) ? s[tid - off] : 0u;
    __syncthreads();
    s[tid] += t;
    __syncthreads();
  }
  unsigned pos = blockOffs[blockIdx.x] + s[tid] - c;  // exclusive offset
#pragma unroll
  for (int j = 0; j < WPT; ++j) {
    unsigned m = w[j];
    unsigned wi = (unsigned)(base + j);
    while (m) {
      int b = __ffs(m) - 1;
      m &= m - 1;
      unsigned key = wi * 32u + (unsigned)b;
      if (pos < (unsigned)U) {
        mkeys[pos] = key;
        out_idx[pos] = (float)(key / (unsigned)N);
        out_idx[(unsigned)U + pos] = (float)(key % (unsigned)N);
      }
      ++pos;
    }
  }
}

// ---------------- scatter edge_attr ----------------
__global__ void k_scatter_edges(const int* __restrict__ e0, const int* __restrict__ e1,
                                const float* __restrict__ edge_attr,
                                const unsigned* __restrict__ mkeys, int U,
                                float* __restrict__ out_val, int E, int N) {
  long long t = (long long)blockIdx.x * blockDim.x + threadIdx.x;
  int i = (int)(t >> 7);
  int d = (int)(t & 127);
  if (i >= E) return;
  unsigned key = (unsigned)e0[i] * (unsigned)N + (unsigned)e1[i];
  int pos = bsearch_key(mkeys, U, key);
  if (pos >= 0)
    unsafeAtomicAdd(&out_val[(size_t)pos * OUT_DIM + d], edge_attr[(size_t)i * OUT_DIM + d]);
}

// ---------------- fused WMMA GEMM + masked scatter ----------------
// One wave computes a 16-row strip of proj = rrwp_val @ W^T via 8x
// v_wmma_f32_16x16x32_f16 (K=32 in one shot), then scatters matched rows
// straight into out_val; proj is never written to memory.
__global__ void k_gemm_scatter(const float* __restrict__ rrwp_val,
                               const int* __restrict__ p0, const int* __restrict__ p1,
                               const float* __restrict__ Wm,
                               const unsigned* __restrict__ mkeys, int U,
                               float* __restrict__ out_val, int R, int N) {
  __shared__ __align__(16) float    Wf32[OUT_DIM * EMB_DIM];  // 16 KB raw W
  __shared__ __align__(16) _Float16 Wh[OUT_DIM * 40];         // padded f16 (80B rows)

  // Stage W into LDS with the CDNA5 async path: 1024 x b128 chunks.
  {
    unsigned ldsbase = (unsigned)(uintptr_t)(&Wf32[0]);
    for (int chunk = threadIdx.x; chunk < 1024; chunk += GEMM_TPB) {
      unsigned lds = ldsbase + (unsigned)chunk * 16u;
      uint64_t g = (uint64_t)(uintptr_t)Wm + (uint64_t)chunk * 16u;
      asm volatile("global_load_async_to_lds_b128 %0, %1, off"
                   :: "v"(lds), "v"(g) : "memory");
    }
    asm volatile("s_wait_asynccnt 0x0" ::: "memory");
  }
  __syncthreads();

  // Convert to f16, padded row stride 40 halves (80B): 16B-aligned rows,
  // no LDS bank conflicts across the 16 lanes reading B fragments.
  for (int e = threadIdx.x; e < OUT_DIM * EMB_DIM; e += GEMM_TPB) {
    int n = e >> 5, k = e & 31;
    Wh[n * 40 + k] = (_Float16)Wf32[e];
  }
  __syncthreads();

  const int lane = threadIdx.x & 31;
  const int wv   = threadIdx.x >> 5;
  const int ln   = lane & 15;
  const int hi   = lane >> 4;
  const int tiles = (R + 15) >> 4;
  const int tile = blockIdx.x * (GEMM_TPB / 32) + wv;
  if (tile >= tiles) return;
  const int m0 = tile << 4;

  // C-fragment rows owned by this lane: m = g + hi*8. Search once per row.
  int pos[8];
  int anyf = 0;
#pragma unroll
  for (int g = 0; g < 8; ++g) {
    int row = m0 + hi * 8 + g;
    if (row < R) {
      unsigned key = (unsigned)p0[row] * (unsigned)N + (unsigned)p1[row];
      pos[g] = bsearch_key(mkeys, U, key);
    } else {
      pos[g] = -1;
    }
    anyf |= (pos[g] >= 0) ? 1 : 0;
  }
  if (__ballot(anyf) == 0ull) return;  // no row of this tile survives the mask

  // A fragment: row m0+ln, halves 0-7 = K hi*8.., halves 8-15 = K 16+hi*8..
  int arow = m0 + ln;
  if (arow >= R) arow = R - 1;
  const float* ap = rrwp_val + (size_t)arow * EMB_DIM + hi * 8;
  float4 f0 = *(const float4*)(ap);
  float4 f1 = *(const float4*)(ap + 4);
  float4 f2 = *(const float4*)(ap + 16);
  float4 f3 = *(const float4*)(ap + 20);
  v16h a;
  a[0] = (_Float16)f0.x;  a[1] = (_Float16)f0.y;  a[2] = (_Float16)f0.z;  a[3] = (_Float16)f0.w;
  a[4] = (_Float16)f1.x;  a[5] = (_Float16)f1.y;  a[6] = (_Float16)f1.z;  a[7] = (_Float16)f1.w;
  a[8] = (_Float16)f2.x;  a[9] = (_Float16)f2.y;  a[10] = (_Float16)f2.z; a[11] = (_Float16)f2.w;
  a[12] = (_Float16)f3.x; a[13] = (_Float16)f3.y; a[14] = (_Float16)f3.z; a[15] = (_Float16)f3.w;

#pragma unroll
  for (int nt = 0; nt < 8; ++nt) {
    // B fragment: N=ln, halves 0-7 = K hi*8.., halves 8-15 = K 16+hi*8..
    // of Wh row (nt*16+ln), matching the A fragment.
    const _Float16* wr = &Wh[(nt * 16 + ln) * 40 + hi * 8];
    v8h b0 = *(const v8h*)(wr);
    v8h b1 = *(const v8h*)(wr + 16);
    v16h b;
#pragma unroll
    for (int i = 0; i < 8; ++i) { b[i] = b0[i]; b[i + 8] = b1[i]; }

    v8f c = {};
    c = __builtin_amdgcn_wmma_f32_16x16x32_f16(
        /*neg_a=*/false, a, /*neg_b=*/false, b,
        /*c_mod=*/(short)0, c, /*reuse_a=*/false, /*reuse_b=*/false);

    int n = nt * 16 + ln;
#pragma unroll
    for (int g = 0; g < 8; ++g) {
      if (pos[g] >= 0)
        unsafeAtomicAdd(&out_val[(size_t)pos[g] * OUT_DIM + n], c[g]);
    }
  }
}

// ---------------- host side ----------------
extern "C" void kernel_launch(void* const* d_in, const int* in_sizes, int n_in,
                              void* d_out, int out_size, void* d_ws, size_t ws_size,
                              hipStream_t stream) {
  const int*   rrwp_idx  = (const int*)d_in[0];    // [2, R]
  const float* rrwp_val  = (const float*)d_in[1];  // [R, 32]
  const int*   edge_idx  = (const int*)d_in[2];    // [2, E]
  const float* edge_attr = (const float*)d_in[3];  // [E, 128]
  const float* Wm        = (const float*)d_in[4];  // [128, 32]

  const int R = in_sizes[0] / 2;
  const int E = in_sizes[2] / 2;
  const int N = N_NODES;
  const int U = out_size / 130;  // out = [2U idx] ++ [U*128 vals]

  const long long NWll = ((long long)N * (long long)N + 31) / 32;  // 50,000,000
  const int NW  = (int)NWll;
  const int NB2 = (NW + EMIT_TPB * WPT - 1) / (EMIT_TPB * WPT);

  unsigned* bitmap    = (unsigned*)d_ws;            // 200 MB, ~L2-resident
  unsigned* blockSums = bitmap + NW;
  unsigned* mkeys     = blockSums + NB2 + 16;

  float* out_idx_f = (float*)d_out;
  float* out_val   = (float*)d_out + (size_t)2 * (size_t)U;

  // 1) zero bitmap (NW divisible by 4)
  {
    long long n4 = NWll / 4;
    int blocks = (int)((n4 + 255) / 256);
    k_zero_u4<<<blocks, 256, 0, stream>>>((uint4*)bitmap, n4);
  }
  // 2) zero out_val (replay-safe: atomics below accumulate)
  {
    long long n = (long long)U * OUT_DIM;
    int blocks = (int)((n + 255) / 256);
    if (blocks > 0) k_zero_f32<<<blocks, 256, 0, stream>>>(out_val, n);
  }
  // 3) set mask bits
  {
    int M = E + N;
    k_set_bits<<<(M + 255) / 256, 256, 0, stream>>>(edge_idx, edge_idx + E, bitmap, E, N);
  }
  // 4) per-block popcounts
  k_blockpop<<<NB2, EMIT_TPB, 0, stream>>>(bitmap, blockSums, NW);
  // 5) exclusive scan of block sums
  k_scan_single<<<1, 1024, 0, stream>>>(blockSums, NB2);
  // 6) emit sorted unique keys + out_idx
  k_emit<<<NB2, EMIT_TPB, 0, stream>>>(bitmap, blockSums, mkeys, out_idx_f, U, NW, N);
  // 7) scatter edge_attr
  {
    long long tot = (long long)E * OUT_DIM;
    int blocks = (int)((tot + 255) / 256);
    k_scatter_edges<<<blocks, 256, 0, stream>>>(edge_idx, edge_idx + E, edge_attr,
                                                mkeys, U, out_val, E, N);
  }
  // 8) fused WMMA GEMM + masked scatter (proj never materialized)
  {
    int tiles = (R + 15) / 16;
    int blocks = (tiles + (GEMM_TPB / 32) - 1) / (GEMM_TPB / 32);
    k_gemm_scatter<<<blocks, GEMM_TPB, 0, stream>>>(rrwp_val, rrwp_idx, rrwp_idx + R,
                                                    Wm, mkeys, U, out_val, R, N);
  }
}
